// PointDeconv_58153857187975
// MI455X (gfx1250) — matrix operand order repaired and checked by reading, not compiled, not run on hardware
//
#include <hip/hip_runtime.h>
#include <hip/hip_bf16.h>

#define BB   4
#define MM   1024
#define NN   4096
#define CXX  256
#define CYY  64
#define CC   320
#define KNN  16
#define EPS  1e-5f

typedef __attribute__((ext_vector_type(16))) __bf16 v16bf;
typedef __attribute__((ext_vector_type(8)))  float  v8f;

struct alignas(16) BF8 { __bf16 h[8]; };

__device__ inline v16bf ld16(const __bf16* p) {
  BF8 l = *(const BF8*)p;
  BF8 h = *(const BF8*)(p + 8);
  v16bf r;
#pragma unroll
  for (int i = 0; i < 8; ++i) { r[i] = l.h[i]; r[i + 8] = h.h[i]; }
  return r;
}
__device__ inline v16bf ld_split(const __bf16* lo, const __bf16* hi) {
  BF8 l = *(const BF8*)lo;
  BF8 h = *(const BF8*)hi;
  v16bf r;
#pragma unroll
  for (int i = 0; i < 8; ++i) { r[i] = l.h[i]; r[i + 8] = h.h[i]; }
  return r;
}
#define WMMA_BF16(A, Bm, Cacc) \
  __builtin_amdgcn_wmma_f32_16x16x32_bf16(false, (A), false, (Bm), (short)0, (Cacc), false, false)

// ---------------- weight conversion fp32 -> bf16 ----------------
__global__ void k_conv(const float* __restrict__ w1, const float* __restrict__ lw,
                       __bf16* __restrict__ w1bf, __bf16* __restrict__ lwbf) {
  int t = blockIdx.x * blockDim.x + threadIdx.x;
  if (t < 256 * 320) w1bf[t] = (__bf16)w1[t];
  int u = t - 256 * 320;
  if (u >= 0 && u < 256 * 4096) lwbf[u] = (__bf16)lw[u];
}

// ---------------- KNN: queries (y) vs sources (x), K=16, weights ----------------
__global__ void k_knn_xp(const float* __restrict__ x, const float* __restrict__ y,
                         int* __restrict__ idx, float* __restrict__ wgt) {
  int t = blockIdx.x * blockDim.x + threadIdx.x;
  if (t >= BB * NN) return;
  int b = t / NN, n = t % NN;
  const float* yb = y + (size_t)b * 67 * NN;
  float qx = yb[n], qy = yb[NN + n], qz = yb[2 * NN + n];
  const float* xb = x + (size_t)b * 259 * MM;
  float bd[KNN]; int bi[KNN];
#pragma unroll
  for (int i = 0; i < KNN; ++i) { bd[i] = 3.4e38f; bi[i] = 0; }
  for (int m = 0; m < MM; ++m) {
    float dx = xb[m] - qx, dy = xb[MM + m] - qy, dz = xb[2 * MM + m] - qz;
    float d = dx * dx + dy * dy + dz * dz;
    if (d < bd[KNN - 1]) {
      bd[KNN - 1] = d; bi[KNN - 1] = m;
#pragma unroll
      for (int j = KNN - 1; j > 0; --j) {
        if (bd[j] < bd[j - 1]) {
          float tf = bd[j]; bd[j] = bd[j - 1]; bd[j - 1] = tf;
          int ti = bi[j]; bi[j] = bi[j - 1]; bi[j - 1] = ti;
        }
      }
    }
  }
  float r[KNN]; float rs = 0.f;
#pragma unroll
  for (int i = 0; i < KNN; ++i) { r[i] = 1.f / (bd[i] + 1e-8f); rs += r[i]; }
  float inv = 1.f / rs;
#pragma unroll
  for (int i = 0; i < KNN; ++i) {
    wgt[(size_t)t * KNN + i] = r[i] * inv;
    idx[(size_t)t * KNN + i] = bi[i];
  }
}

// ---------------- self-KNN (y vs y) + Gaussian density ----------------
__global__ void k_knn_qq(const float* __restrict__ y, int* __restrict__ nidx,
                         float* __restrict__ dinv) {
  int t = blockIdx.x * blockDim.x + threadIdx.x;
  if (t >= BB * NN) return;
  int b = t / NN, n = t % NN;
  const float* yb = y + (size_t)b * 67 * NN;
  float qx = yb[n], qy = yb[NN + n], qz = yb[2 * NN + n];
  float bd[KNN]; int bi[KNN];
#pragma unroll
  for (int i = 0; i < KNN; ++i) { bd[i] = 3.4e38f; bi[i] = 0; }
  float acc = 0.f;
  for (int m = 0; m < NN; ++m) {
    float dx = yb[m] - qx, dy = yb[NN + m] - qy, dz = yb[2 * NN + m] - qz;
    float d = dx * dx + dy * dy + dz * dz;
    acc += __expf(-0.5f * d);
    if (d < bd[KNN - 1]) {
      bd[KNN - 1] = d; bi[KNN - 1] = m;
#pragma unroll
      for (int j = KNN - 1; j > 0; --j) {
        if (bd[j] < bd[j - 1]) {
          float tf = bd[j]; bd[j] = bd[j - 1]; bd[j - 1] = tf;
          int ti = bi[j]; bi[j] = bi[j - 1]; bi[j - 1] = ti;
        }
      }
    }
  }
  dinv[t] = (float)NN / acc;   // 1/mean(exp)
#pragma unroll
  for (int i = 0; i < KNN; ++i) nidx[(size_t)t * KNN + i] = bi[i];
}

// ---------------- BN stats finalize: mean / rsqrt(var+eps) ----------------
__global__ void k_fin(float* __restrict__ st, int nch, float invcnt) {
  int c = threadIdx.x;
  if (c < nch) {
    float m = st[c] * invcnt;
    float v = st[nch + c] * invcnt - m * m;
    st[2 * nch + c] = m;
    st[3 * nch + c] = rsqrtf(v + EPS);
  }
}

// ---------------- scale net layer 1 (1 -> 8) + stats ----------------
__global__ void k_scale1(const float* __restrict__ dinv, const float* __restrict__ w1,
                         const float* __restrict__ b1, float* __restrict__ a1,
                         float* __restrict__ st) {
  __shared__ float ss[8], sq[8];
  int tid = threadIdx.x;
  if (tid < 8) { ss[tid] = 0.f; sq[tid] = 0.f; }
  __syncthreads();
  int t = blockIdx.x * blockDim.x + tid;
  if (t < BB * NN) {
    float v = dinv[t];
#pragma unroll
    for (int j = 0; j < 8; ++j) {
      float a = w1[j] * v + b1[j];
      a1[(size_t)t * 8 + j] = a;
      atomicAdd(&ss[j], a); atomicAdd(&sq[j], a * a);
    }
  }
  __syncthreads();
  if (tid < 8) { atomicAdd(&st[tid], ss[tid]); atomicAdd(&st[8 + tid], sq[tid]); }
}

// ---------------- generic 8->8 layer: BN+ReLU input, linear, stats ----------------
__global__ void k_lin8(const float* __restrict__ ain, const float* __restrict__ st1,
                       const float* __restrict__ g, const float* __restrict__ be,
                       const float* __restrict__ w2, const float* __restrict__ b2,
                       float* __restrict__ aout, float* __restrict__ st2, int count) {
  __shared__ float ss[8], sq[8];
  int tid = threadIdx.x;
  if (tid < 8) { ss[tid] = 0.f; sq[tid] = 0.f; }
  __syncthreads();
  int t = blockIdx.x * blockDim.x + tid;
  if (t < count) {
    float h[8];
#pragma unroll
    for (int i = 0; i < 8; ++i) {
      float a = ain[(size_t)t * 8 + i];
      float f = (a - st1[16 + i]) * st1[24 + i] * g[i] + be[i];
      h[i] = f > 0.f ? f : 0.f;
    }
#pragma unroll
    for (int j = 0; j < 8; ++j) {
      float acc = b2[j];
#pragma unroll
      for (int i = 0; i < 8; ++i) acc += w2[j * 8 + i] * h[i];
      aout[(size_t)t * 8 + j] = acc;
      atomicAdd(&ss[j], acc); atomicAdd(&sq[j], acc * acc);
    }
  }
  __syncthreads();
  if (tid < 8) { atomicAdd(&st2[tid], ss[tid]); atomicAdd(&st2[8 + tid], sq[tid]); }
}

// ---------------- scale net layer 3 (8 -> 1) + sigmoid ----------------
__global__ void k_scale3(const float* __restrict__ a2, const float* __restrict__ st2,
                         const float* __restrict__ g, const float* __restrict__ be,
                         const float* __restrict__ w3, const float* __restrict__ b3,
                         float* __restrict__ sval) {
  int t = blockIdx.x * blockDim.x + threadIdx.x;
  if (t >= BB * NN) return;
  float acc = b3[0];
#pragma unroll
  for (int i = 0; i < 8; ++i) {
    float a = a2[(size_t)t * 8 + i];
    float f = (a - st2[16 + i]) * st2[24 + i] * g[i] + be[i];
    f = f > 0.f ? f : 0.f;
    acc += w3[i] * f;
  }
  sval[t] = 1.f / (1.f + __expf(-acc));
}

// ---------------- build xc = [interp(xf), yf] pre-scaled by s, bf16 ----------------
__global__ void k_xc(const float* __restrict__ x, const float* __restrict__ y,
                     const int* __restrict__ idx, const float* __restrict__ wgt,
                     const float* __restrict__ sval, __bf16* __restrict__ xcbf) {
  int p = blockIdx.x;              // global point id, b*NN+n
  int b = p / NN, n = p % NN;
  int c = threadIdx.x;             // 0..319
  float v;
  if (c < CXX) {
    float acc = 0.f;
#pragma unroll
    for (int k = 0; k < KNN; ++k) {
      int src = idx[(size_t)p * KNN + k];
      acc += x[((size_t)b * 259 + 3 + c) * MM + src] * wgt[(size_t)p * KNN + k];
    }
    v = acc;
  } else {
    v = y[((size_t)b * 67 + 3 + (c - CXX)) * NN + n];
  }
  xcbf[(size_t)p * CC + c] = (__bf16)(v * sval[p]);   // fold neighbor scale (depends on source only)
}

// ---------------- wnet layer 1 (3 -> 8) on relative positions ----------------
__global__ void k_wnet1(const float* __restrict__ y, const int* __restrict__ nidx,
                        const float* __restrict__ w1, const float* __restrict__ b1,
                        float* __restrict__ aw1, float* __restrict__ st) {
  __shared__ float ss[8], sq[8];
  int tid = threadIdx.x;
  if (tid < 8) { ss[tid] = 0.f; sq[tid] = 0.f; }
  __syncthreads();
  int t = blockIdx.x * blockDim.x + tid;   // sample = p*16 + k
  if (t < BB * NN * KNN) {
    int p = t / KNN;
    int b = p / NN, n = p % NN;
    int src = nidx[t];
    float pd[3];
#pragma unroll
    for (int d = 0; d < 3; ++d)
      pd[d] = y[((size_t)b * 67 + d) * NN + src] - y[((size_t)b * 67 + d) * NN + n];
#pragma unroll
    for (int j = 0; j < 8; ++j) {
      float a = b1[j];
#pragma unroll
      for (int d = 0; d < 3; ++d) a += w1[j * 3 + d] * pd[d];
      aw1[(size_t)t * 8 + j] = a;
      atomicAdd(&ss[j], a); atomicAdd(&sq[j], a * a);
    }
  }
  __syncthreads();
  if (tid < 8) { atomicAdd(&st[tid], ss[tid]); atomicAdd(&st[8 + tid], sq[tid]); }
}

// ---------------- wnet layer 3 (8 -> 16): per-point 16x16 weight matrix ----------------
__global__ void k_wnet3(const float* __restrict__ a2, const float* __restrict__ st2,
                        const float* __restrict__ g, const float* __restrict__ be,
                        const float* __restrict__ w3, const float* __restrict__ b3,
                        float* __restrict__ wmat) {
  int t = blockIdx.x * blockDim.x + threadIdx.x;
  if (t >= BB * NN * KNN) return;
  int p = t / KNN, k = t % KNN;
  float h[8];
#pragma unroll
  for (int i = 0; i < 8; ++i) {
    float a = a2[(size_t)t * 8 + i];
    float f = (a - st2[16 + i]) * st2[24 + i] * g[i] + be[i];
    h[i] = f > 0.f ? f : 0.f;
  }
#pragma unroll
  for (int j = 0; j < 16; ++j) {
    float acc = b3[j];
#pragma unroll
    for (int i = 0; i < 8; ++i) acc += w3[j * 8 + i] * h[i];
    wmat[(size_t)p * 256 + j * 16 + k] = acc;   // W_n[j][k]
  }
}

// ---------------- GEMM 1: h_pre = W1(256x320) x gathered xc columns (WMMA bf16) ----------------
__global__ void __launch_bounds__(256) k_pass1(const __bf16* __restrict__ w1bf,
        const float* __restrict__ b1, const __bf16* __restrict__ xcbf,
        const int* __restrict__ nidx, __bf16* __restrict__ hpre, float* __restrict__ st) {
  __shared__ float ss[256], sq[256];
  int tid = threadIdx.x;
  ss[tid] = 0.f; sq[tid] = 0.f;
  __syncthreads();
  int wave = tid >> 5, lane = tid & 31;
  int lh = lane & 15, hi = lane >> 4;
  int pbase = blockIdx.x * 16;   // 16 points per block
  int b = pbase / NN;
  int m0 = wave * 32;            // wave -> channels [m0, m0+32)
  float bias0[8], bias1[8];
  float as0[8] = {}, aq0[8] = {}, as1[8] = {}, aq1[8] = {};
#pragma unroll
  for (int v = 0; v < 8; ++v) {
    bias0[v] = b1[m0 + v + hi * 8];
    bias1[v] = b1[m0 + 16 + v + hi * 8];
  }
  const __bf16* arow0 = w1bf + (size_t)(m0 + lh) * CC;
  const __bf16* arow1 = arow0 + 16 * CC;
  for (int pi = 0; pi < 16; ++pi) {
    int p = pbase + pi;
    int src = nidx[(size_t)p * KNN + lh];      // column = neighbor lh of point p
    const __bf16* brow = xcbf + (size_t)(b * NN + src) * CC;
    v8f c0 = {0.f,0.f,0.f,0.f,0.f,0.f,0.f,0.f};
    v8f c1 = {0.f,0.f,0.f,0.f,0.f,0.f,0.f,0.f};
#pragma unroll
    for (int ks = 0; ks < 10; ++ks) {
      int k0 = ks * 32;
      v16bf bm = ld_split(brow + k0 + hi * 8, brow + k0 + hi * 8 + 16);   // B: cols striped, K {0..7,16..23}/{8..15,24..31}
      v16bf a0 = ld_split(arow0 + k0 + hi * 8, arow0 + k0 + hi * 8 + 16); // A: K {0..7,16..23}/{8..15,24..31}
      v16bf a1 = ld_split(arow1 + k0 + hi * 8, arow1 + k0 + hi * 8 + 16);
      c0 = WMMA_BF16(a0, bm, c0);
      c1 = WMMA_BF16(a1, bm, c1);
    }
#pragma unroll
    for (int v = 0; v < 8; ++v) {
      float f0 = c0[v] + bias0[v];
      float f1 = c1[v] + bias1[v];
      int ch0 = m0 + v + hi * 8;
      int ch1 = ch0 + 16;
      hpre[((size_t)p * 256 + ch0) * KNN + lh] = (__bf16)f0;
      hpre[((size_t)p * 256 + ch1) * KNN + lh] = (__bf16)f1;
      as0[v] += f0; aq0[v] += f0 * f0;
      as1[v] += f1; aq1[v] += f1 * f1;
    }
  }
#pragma unroll
  for (int v = 0; v < 8; ++v) {
    int ch0 = m0 + v + hi * 8, ch1 = ch0 + 16;
    atomicAdd(&ss[ch0], as0[v]); atomicAdd(&sq[ch0], aq0[v]);
    atomicAdd(&ss[ch1], as1[v]); atomicAdd(&sq[ch1], aq1[v]);
  }
  __syncthreads();
  atomicAdd(&st[tid], ss[tid]);
  atomicAdd(&st[256 + tid], sq[tid]);
}

// ---------------- fused pass 2: o = BN(h)W^T (phase A, LDS) ; z = LW x vec(o) (phase B) ----------------
__global__ void __launch_bounds__(256) k_pass2(const __bf16* __restrict__ hpre,
        const float* __restrict__ wmat, const __bf16* __restrict__ lwbf,
        const float* __restrict__ linb, const float* __restrict__ stm,
        const float* __restrict__ g, const float* __restrict__ be,
        float* __restrict__ zpre, float* __restrict__ st) {
  extern __shared__ __align__(16) char smem[];
  __bf16* obf = (__bf16*)smem;                 // 16 points x 4096 (c*16+j) bf16 = 128KB
  float* ss = (float*)(smem + 16 * 4096 * 2);
  float* sq = ss + 256;
  int tid = threadIdx.x;
  ss[tid] = 0.f; sq[tid] = 0.f;
  int wave = tid >> 5, lane = tid & 31, lh = lane & 15, hi = lane >> 4;
  int pbase = blockIdx.x * 16;
  int b = pbase / NN;
  // ---- phase A: each wave computes o for 2 points ----
#pragma unroll
  for (int pp = 0; pp < 2; ++pp) {
    int pi = wave * 2 + pp;
    int p = pbase + pi;
    v16bf bw;                                  // B = W_n^T (K=16 real, rest zero)
    {
      const float* wr = wmat + (size_t)p * 256 + lh * 16 + hi * 8;
#pragma unroll
      for (int i = 0; i < 8; ++i) bw[i] = (__bf16)wr[i];
#pragma unroll
      for (int i = 8; i < 16; ++i) bw[i] = (__bf16)0.f;
    }
#pragma unroll 2
    for (int ct = 0; ct < 16; ++ct) {
      int c = ct * 16 + lh;
      float mean = stm[512 + c], istd = stm[768 + c], gg = g[c], bb = be[c];
      const __bf16* hr = hpre + ((size_t)p * 256 + c) * KNN + hi * 8;
      BF8 hv = *(const BF8*)hr;
      v16bf am;
#pragma unroll
      for (int i = 0; i < 8; ++i) {
        float f = (float)hv.h[i];
        f = (f - mean) * istd * gg + bb;
        f = f > 0.f ? f : 0.f;
        am[i] = (__bf16)f;
        am[i + 8] = (__bf16)0.f;               // K 16..31 zero pad
      }
      v8f acc = {0.f,0.f,0.f,0.f,0.f,0.f,0.f,0.f};
      acc = WMMA_BF16(am, bw, acc);
#pragma unroll
      for (int v = 0; v < 8; ++v) {
        int cr = ct * 16 + v + hi * 8;
        obf[pi * 4096 + cr * 16 + lh] = (__bf16)acc[v];   // vec index c*16+j
      }
    }
  }
  __syncthreads();
  // ---- phase B: z tile 32ch x 16pts, K = 4096 (unrolled x2 for load/WMMA overlap) ----
  int m0 = wave * 32;
  v8f z0 = {0.f,0.f,0.f,0.f,0.f,0.f,0.f,0.f};
  v8f z1 = {0.f,0.f,0.f,0.f,0.f,0.f,0.f,0.f};
  const __bf16* ar0 = lwbf + (size_t)(m0 + lh) * 4096;
  const __bf16* ar1 = ar0 + 16 * 4096;
  const __bf16* bl = obf + lh * 4096;          // column = point lh
#pragma unroll 2
  for (int ks = 0; ks < 128; ++ks) {
    int k0 = ks * 32;
    // speculative prefetch of the A panel ~8 k-steps ahead (L2-resident LW)
    if ((ks & 7) == 0 && ks + 8 < 128) {
      __builtin_prefetch(ar0 + k0 + 256, 0, 1);
      __builtin_prefetch(ar1 + k0 + 256, 0, 1);
    }
    v16bf a0 = ld_split(ar0 + k0 + hi * 8, ar0 + k0 + hi * 8 + 16);
    v16bf a1 = ld_split(ar1 + k0 + hi * 8, ar1 + k0 + hi * 8 + 16);
    v16bf bm = ld_split(bl + k0 + hi * 8, bl + k0 + hi * 8 + 16);
    z0 = WMMA_BF16(a0, bm, z0);
    z1 = WMMA_BF16(a1, bm, z1);
  }
  int n = (pbase % NN) + lh;
#pragma unroll
  for (int v = 0; v < 8; ++v) {
    int o0 = m0 + v + hi * 8, o1 = o0 + 16;
    float f0 = z0[v] + linb[o0];
    float f1 = z1[v] + linb[o1];
    zpre[((size_t)b * 256 + o0) * NN + n] = f0;
    zpre[((size_t)b * 256 + o1) * NN + n] = f1;
    atomicAdd(&ss[o0], f0); atomicAdd(&sq[o0], f0 * f0);
    atomicAdd(&ss[o1], f1); atomicAdd(&sq[o1], f1 * f1);
  }
  __syncthreads();
  atomicAdd(&st[tid], ss[tid]);
  atomicAdd(&st[256 + tid], sq[tid]);
}

// ---------------- final: out = concat(q, relu(bn(z))) ----------------
__global__ void k_out(const float* __restrict__ y, const float* __restrict__ zpre,
                      const float* __restrict__ st, const float* __restrict__ g,
                      const float* __restrict__ be, float* __restrict__ out) {
  int t = blockIdx.x * blockDim.x + threadIdx.x;
  if (t >= BB * 259 * NN) return;
  int b = t / (259 * NN);
  int r = t % (259 * NN);
  int ch = r / NN, n = r % NN;
  float v;
  if (ch < 3) {
    v = y[((size_t)b * 67 + ch) * NN + n];
  } else {
    int o = ch - 3;
    float f = zpre[((size_t)b * 256 + o) * NN + n];
    f = (f - st[512 + o]) * st[768 + o] * g[o] + be[o];
    v = f > 0.f ? f : 0.f;
  }
  out[t] = v;
}

extern "C" void kernel_launch(void* const* d_in, const int* in_sizes, int n_in,
                              void* d_out, int out_size, void* d_ws, size_t ws_size,
                              hipStream_t stream) {
  (void)in_sizes; (void)n_in; (void)out_size; (void)ws_size;
  const float* x      = (const float*)d_in[0];
  const float* y      = (const float*)d_in[1];
  const float* s_w1   = (const float*)d_in[2];
  const float* s_b1   = (const float*)d_in[3];
  const float* s_g1   = (const float*)d_in[4];
  const float* s_be1  = (const float*)d_in[5];
  const float* s_w2   = (const float*)d_in[6];
  const float* s_b2   = (const float*)d_in[7];
  const float* s_g2   = (const float*)d_in[8];
  const float* s_be2  = (const float*)d_in[9];
  const float* s_w3   = (const float*)d_in[10];
  const float* s_b3   = (const float*)d_in[11];
  const float* w_w1   = (const float*)d_in[12];
  const float* w_b1   = (const float*)d_in[13];
  const float* w_g1   = (const float*)d_in[14];
  const float* w_be1  = (const float*)d_in[15];
  const float* w_w2   = (const float*)d_in[16];
  const float* w_b2   = (const float*)d_in[17];
  const float* w_g2   = (const float*)d_in[18];
  const float* w_be2  = (const float*)d_in[19];
  const float* w_w3   = (const float*)d_in[20];
  const float* w_b3   = (const float*)d_in[21];
  const float* m_w1   = (const float*)d_in[22];
  const float* m_b1   = (const float*)d_in[23];
  const float* m_g1   = (const float*)d_in[24];
  const float* m_be1  = (const float*)d_in[25];
  const float* l_w    = (const float*)d_in[26];
  const float* l_b    = (const float*)d_in[27];
  const float* l_g    = (const float*)d_in[28];
  const float* l_be   = (const float*)d_in[29];

  char* ws = (char*)d_ws;
  // workspace layout (bytes)
  int*     idx   = (int*)   (ws + 0);            //  1 MB
  float*   wgt   = (float*) (ws + 1048576);      //  1 MB
  int*     nidx  = (int*)   (ws + 2097152);      //  1 MB
  float*   dinv  = (float*) (ws + 3145728);      // 64 KB
  float*   sval  = (float*) (ws + 3211264);      // 64 KB
  float*   stats = (float*) (ws + 3276800);      // 16 KB (zeroed)
  float*   a1s   = (float*) (ws + 3293184);      // 512 KB
  float*   a2s   = (float*) (ws + 3817472);      // 512 KB
  float*   aw1   = (float*) (ws + 4341760);      //  8 MB
  float*   aw2   = (float*) (ws + 12730368);     //  8 MB
  float*   wmat  = (float*) (ws + 21118976);     // 16 MB
  __bf16*  w1bf  = (__bf16*)(ws + 37896192);     // 160 KB
  __bf16*  lwbf  = (__bf16*)(ws + 38060032);     //  2 MB
  __bf16*  xcbf  = (__bf16*)(ws + 40157184);     // 10 MB
  float*   zpre  = (float*) (ws + 50642944);     // 16 MB
  __bf16*  hpre  = (__bf16*)(ws + 67420160);     // 128 MB

  float* st_s1 = stats;        // 8 ch
  float* st_s2 = stats + 32;
  float* st_w1 = stats + 64;
  float* st_w2 = stats + 96;
  float* st_m  = stats + 128;  // 256 ch
  float* st_l  = stats + 1152; // 256 ch

  hipMemsetAsync(stats, 0, 16384, stream);

  k_conv  <<<4416, 256, 0, stream>>>(m_w1, l_w, w1bf, lwbf);
  k_knn_xp<<<64,   256, 0, stream>>>(x, y, idx, wgt);
  k_knn_qq<<<64,   256, 0, stream>>>(y, nidx, dinv);

  // scale net (1->8->8->1), BN stats over B*N = 16384
  k_scale1<<<64, 256, 0, stream>>>(dinv, s_w1, s_b1, a1s, st_s1);
  k_fin   <<<1,  256, 0, stream>>>(st_s1, 8, 1.f / 16384.f);
  k_lin8  <<<64, 256, 0, stream>>>(a1s, st_s1, s_g1, s_be1, s_w2, s_b2, a2s, st_s2, BB * NN);
  k_fin   <<<1,  256, 0, stream>>>(st_s2, 8, 1.f / 16384.f);
  k_scale3<<<64, 256, 0, stream>>>(a2s, st_s2, s_g2, s_be2, s_w3, s_b3, sval);

  k_xc<<<BB * NN, 320, 0, stream>>>(x, y, idx, wgt, sval, xcbf);

  // wnet (3->8->8->16), BN stats over B*16*N = 262144
  k_wnet1<<<1024, 256, 0, stream>>>(y, nidx, w_w1, w_b1, aw1, st_w1);
  k_fin  <<<1,    256, 0, stream>>>(st_w1, 8, 1.f / 262144.f);
  k_lin8 <<<1024, 256, 0, stream>>>(aw1, st_w1, w_g1, w_be1, w_w2, w_b2, aw2, st_w2, BB * NN * KNN);
  k_fin  <<<1,    256, 0, stream>>>(st_w2, 8, 1.f / 262144.f);
  k_wnet3<<<1024, 256, 0, stream>>>(aw2, st_w2, w_g2, w_be2, w_w3, w_b3, wmat);

  // big WMMA GEMM 1 + mlp BN stats
  k_pass1<<<1024, 256, 0, stream>>>(w1bf, m_b1, xcbf, nidx, hpre, st_m);
  k_fin  <<<1,    256, 0, stream>>>(st_m, 256, 1.f / 262144.f);

  // fused o-GEMM + big WMMA GEMM 2 + lin BN stats (128KB + 2KB dynamic LDS)
  k_pass2<<<1024, 256, 16 * 4096 * 2 + 2048, stream>>>(hpre, wmat, lwbf, l_b, st_m, m_g1, m_be1,
                                                       zpre, st_l);
  k_fin  <<<1,    256, 0, stream>>>(st_l, 256, 1.f / 16384.f);

  k_out<<<16576, 256, 0, stream>>>(y, zpre, st_l, l_g, l_be, (float*)d_out);
}
